// Mamba_14396730376916
// MI455X (gfx1250) — matrix operand order repaired and checked by reading, not compiled, not run on hardware
//
#include <hip/hip_runtime.h>

// ---------------- problem constants ----------------
constexpr int cV  = 50280;
constexpr int cD  = 1024;
constexpr int cE  = 2048;
constexpr int cN  = 16;
constexpr int cDd = 64;
constexpr int cDc = 4;
constexpr int cNL = 2;
constexpr int cB  = 2;
constexpr int cL  = 1024;
constexpr int cT  = cB * cL;   // 2048 tokens
constexpr float cEPS = 1e-5f;

// ---------------- CDNA5 WMMA types ----------------
typedef __attribute__((ext_vector_type(16))) __bf16   v16bf;
typedef __attribute__((ext_vector_type(8)))  float    v8f;
typedef __attribute__((ext_vector_type(4)))  unsigned u32x4;

__device__ __forceinline__ unsigned short f2bfbits(float f) {
    unsigned u = __float_as_uint(f);
    unsigned r = u + 0x7FFFu + ((u >> 16) & 1u);   // round-to-nearest-even
    return (unsigned short)(r >> 16);
}

// =====================================================================
// bf16 WMMA GEMM:  C[M,N] (f32, ldc) = A[M,K] * Bw[N,K]^T  (+C if accum)
// Requirements (met by all call sites): M % 16 == 0, K % 64 == 0.
// Block = 256 threads = 8 waves (2 x 4); wave tile = 32M x 64N;
// block tile = 64M x 256N. Fully branch-free double-buffered K-loop
// (prologue + peeled epilogue) so the backend can keep 12 loads in
// flight across each 8-WMMA burst instead of s_wait_loadcnt 0.
// OOB N columns clamp-load; store-side guard discards them.
// =====================================================================
__global__ __launch_bounds__(256) void wmma_gemm_bf16(
    const unsigned short* __restrict__ A,
    const unsigned short* __restrict__ Bw,
    float* __restrict__ C,
    int M, int N, int K, int ldc, int accum)
{
    const int lane = threadIdx.x & 31;
    const int wave = threadIdx.x >> 5;
    const int wrow = wave >> 2;              // 0..1
    const int wcol = wave & 3;               // 0..3
    const int m0   = blockIdx.x * 64 + wrow * 32;
    const int n0   = blockIdx.y * 256 + wcol * 64;
    if (m0 >= M) return;                     // wave-uniform

    const int half = lane >> 4;              // 0/1
    const int l16  = lane & 15;

    v8f acc[2][4];
#pragma unroll
    for (int mi = 0; mi < 2; ++mi)
#pragma unroll
        for (int j = 0; j < 4; ++j) {
            if (accum) {
#pragma unroll
                for (int r = 0; r < 8; ++r) {
                    int m = m0 + mi * 16 + r + 8 * half;
                    int n = n0 + j * 16 + l16;
                    acc[mi][j][r] = (m < M && n < N) ? C[(long)m * ldc + n] : 0.0f;
                }
            } else {
#pragma unroll
                for (int r = 0; r < 8; ++r) acc[mi][j][r] = 0.0f;
            }
        }

    // A rows (16-bit A 16x32 layout: both lane halves read row l&15,
    // different K sub-ranges).  M%16==0 at all call sites -> in range.
    const unsigned short* __restrict__ arow0 = A + (long)(m0 + l16) * K;
    const unsigned short* __restrict__ arow1 = A + (long)(m0 + 16 + l16) * K;

    // B rows: clamp out-of-range columns (their outputs are never stored;
    // D column n depends only on B column n, which is lane-striped).
    const unsigned short* __restrict__ brow[4];
#pragma unroll
    for (int j = 0; j < 4; ++j) {
        int n  = n0 + j * 16 + l16;
        int nc = (n < N) ? n : (N - 1);
        brow[j] = Bw + (long)nc * K + half * 8;
    }

    union AF { v16bf v; u32x4 q[2]; };
    AF a[2][2];    // [buf][mi]
    AF bq[2][4];   // [buf][j]

    const int kbh = half * 8;

    auto loadA = [&](AF* dst, int k0) {
        dst[0].q[0] = *(const u32x4*)(arow0 + k0 + kbh);
        dst[0].q[1] = *(const u32x4*)(arow0 + k0 + kbh + 16);
        dst[1].q[0] = *(const u32x4*)(arow1 + k0 + kbh);
        dst[1].q[1] = *(const u32x4*)(arow1 + k0 + kbh + 16);
    };
    auto loadB = [&](AF* dst, int k0) {
#pragma unroll
        for (int j = 0; j < 4; ++j) {
            dst[j].q[0] = *(const u32x4*)(brow[j] + k0);
            dst[j].q[1] = *(const u32x4*)(brow[j] + k0 + 16);
        }
    };
    auto mma = [&](AF* av, AF* bv) {
#pragma unroll
        for (int mi = 0; mi < 2; ++mi)
#pragma unroll
            for (int j = 0; j < 4; ++j)
                acc[mi][j] = __builtin_amdgcn_wmma_f32_16x16x32_bf16(
                    false, av[mi].v, false, bv[j].v, (short)0, acc[mi][j],
                    false, false);
    };

    // prologue: fill both buffers (K >= 64 guaranteed)
    loadA(a[0], 0);  loadB(bq[0], 0);
    loadA(a[1], 32); loadB(bq[1], 32);

    // branch-free steady state; last 64-K step peeled
    for (int k0 = 0; k0 < K - 64; k0 += 64) {
        __builtin_prefetch(arow0 + k0 + 512, 0, 1);     // speculative; OOB dropped
        __builtin_prefetch(brow[0] + k0 + 512, 0, 1);
        __builtin_prefetch(brow[2] + k0 + 512, 0, 1);
        mma(a[0], bq[0]);
        loadA(a[0], k0 + 64); loadB(bq[0], k0 + 64);
        mma(a[1], bq[1]);
        loadA(a[1], k0 + 96); loadB(bq[1], k0 + 96);
    }
    mma(a[0], bq[0]);
    mma(a[1], bq[1]);

#pragma unroll
    for (int mi = 0; mi < 2; ++mi)
#pragma unroll
        for (int j = 0; j < 4; ++j) {
            const int n = n0 + j * 16 + l16;
            if (n >= N) continue;
#pragma unroll
            for (int r = 0; r < 8; ++r) {
                const int m = m0 + mi * 16 + r + 8 * half;
                if (m < M) C[(long)m * ldc + n] = acc[mi][j][r];
            }
        }
}

// =====================================================================
// Elementwise / helper kernels
// =====================================================================
__global__ void cvt_bf16_k(const float* __restrict__ src,
                           unsigned short* __restrict__ dst, int n)
{
    int i = blockIdx.x * blockDim.x + threadIdx.x;
    if (i < n) dst[i] = f2bfbits(src[i]);
}

__global__ void embed_k(const int* __restrict__ ids,
                        const float* __restrict__ emb,
                        float* __restrict__ resid)
{
    int i = blockIdx.x * blockDim.x + threadIdx.x;
    if (i >= cT * cD) return;
    int t = i / cD, d = i - t * cD;
    resid[i] = emb[(long)ids[t] * cD + d];
}

// block-per-token RMSNorm, writes bf16 output for the following GEMMs
__global__ __launch_bounds__(256) void rmsnorm_k(
    const float* __restrict__ x, const float* __restrict__ w,
    unsigned short* __restrict__ ybf)
{
    __shared__ float red[8];
    __shared__ float scale;
    const int t = blockIdx.x;
    const float* __restrict__ xr = x + (long)t * cD;

    float s = 0.0f;
    for (int d = threadIdx.x; d < cD; d += 256) { float v = xr[d]; s += v * v; }
#pragma unroll
    for (int off = 16; off > 0; off >>= 1) s += __shfl_down(s, off, 32);
    if ((threadIdx.x & 31) == 0) red[threadIdx.x >> 5] = s;
    __syncthreads();
    if (threadIdx.x == 0) {
        float tot = 0.0f;
#pragma unroll
        for (int i = 0; i < 8; ++i) tot += red[i];
        scale = rsqrtf(tot / (float)cD + cEPS);
    }
    __syncthreads();
    const float sc = scale;
    for (int d = threadIdx.x; d < cD; d += 256)
        ybf[(long)t * cD + d] = f2bfbits(xr[d] * sc * w[d]);
}

// depthwise causal conv (Dc=4) + bias + SiLU; writes f32 and bf16
__global__ void conv_silu_k(const float* __restrict__ xin,
                            const float* __restrict__ cw,
                            const float* __restrict__ cb,
                            float* __restrict__ xs,
                            unsigned short* __restrict__ xsbf)
{
    int i = blockIdx.x * blockDim.x + threadIdx.x;
    if (i >= cT * cE) return;
    int e = i % cE, t = i / cE, b = t / cL, tl = t - b * cL;
    float acc = cb[e];
#pragma unroll
    for (int h = 0; h < cDc; ++h) {
        int src = tl - (cDc - 1) + h;        // tap h multiplies x[t-3+h]
        if (src >= 0) acc += cw[e * cDc + h] * xin[((long)b * cL + src) * cE + e];
    }
    float s = acc / (1.0f + __expf(-acc));   // SiLU
    xs[i] = s;
    xsbf[i] = f2bfbits(s);
}

__global__ void bias_softplus_k(float* __restrict__ delta,
                                const float* __restrict__ bias)
{
    int i = blockIdx.x * blockDim.x + threadIdx.x;
    if (i >= cT * cE) return;
    int e = i % cE;
    float v = delta[i] + bias[e];
    delta[i] = (v > 20.0f) ? v : log1pf(__expf(v));
}

// selective scan: one lane per (b,e) channel; N=16 state kept in registers
__global__ __launch_bounds__(256) void scan_k(
    const float* __restrict__ delta, const float* __restrict__ xs,
    const float* __restrict__ Bt,    const float* __restrict__ Ct,
    const float* __restrict__ A_log, float* __restrict__ y)
{
    int idx = blockIdx.x * blockDim.x + threadIdx.x;
    if (idx >= cB * cE) return;
    const int e = idx % cE;
    const int b = idx / cE;

    float An[cN], h[cN];
#pragma unroll
    for (int n = 0; n < cN; ++n) { An[n] = -__expf(A_log[e * cN + n]); h[n] = 0.0f; }

    for (int tl = 0; tl < cL; ++tl) {
        const long t = (long)b * cL + tl;
        const float dt = delta[t * cE + e];
        const float dx = dt * xs[t * cE + e];
        const float4* __restrict__ Bp = (const float4*)(Bt + t * cN);
        const float4* __restrict__ Cp = (const float4*)(Ct + t * cN);
        float acc = 0.0f;
#pragma unroll
        for (int q = 0; q < 4; ++q) {
            float4 bv = Bp[q], cv = Cp[q];
            h[4*q+0] = __expf(dt * An[4*q+0]) * h[4*q+0] + dx * bv.x; acc += h[4*q+0] * cv.x;
            h[4*q+1] = __expf(dt * An[4*q+1]) * h[4*q+1] + dx * bv.y; acc += h[4*q+1] * cv.y;
            h[4*q+2] = __expf(dt * An[4*q+2]) * h[4*q+2] + dx * bv.z; acc += h[4*q+2] * cv.z;
            h[4*q+3] = __expf(dt * An[4*q+3]) * h[4*q+3] + dx * bv.w; acc += h[4*q+3] * cv.w;
        }
        y[t * cE + e] = acc;
    }
}

// z = (y + xs*WD) * silu(skip), emitted as bf16 for the out-projection GEMM
__global__ void zgate_k(const float* __restrict__ yv,
                        const float* __restrict__ xs,
                        const float* __restrict__ skip,
                        const float* __restrict__ WD,
                        unsigned short* __restrict__ zbf)
{
    int i = blockIdx.x * blockDim.x + threadIdx.x;
    if (i >= cT * cE) return;
    int e = i % cE;
    float s = skip[i];
    float g = s / (1.0f + __expf(-s));
    zbf[i] = f2bfbits((yv[i] + xs[i] * WD[e]) * g);
}

// =====================================================================
// Host-side orchestration
// =====================================================================
extern "C" void kernel_launch(void* const* d_in, const int* in_sizes, int n_in,
                              void* d_out, int out_size, void* d_ws, size_t ws_size,
                              hipStream_t stream)
{
    const int*   ids      = (const int*)  d_in[0];
    const float* emb      = (const float*)d_in[1];
    const float* lm_head  = (const float*)d_in[2];
    const float* norm_w   = (const float*)d_in[3];
    const float* l_norm_w = (const float*)d_in[4];
    const float* l_skip_w = (const float*)d_in[5];
    const float* l_in_w   = (const float*)d_in[6];
    const float* l_conv_w = (const float*)d_in[7];
    const float* l_conv_b = (const float*)d_in[8];
    const float* l_Wd1    = (const float*)d_in[9];
    const float* l_Wd2_w  = (const float*)d_in[10];
    const float* l_Wd2_b  = (const float*)d_in[11];
    const float* l_WB     = (const float*)d_in[12];
    const float* l_WC     = (const float*)d_in[13];
    const float* l_A_log  = (const float*)d_in[14];
    const float* l_WD     = (const float*)d_in[15];
    const float* l_out_w  = (const float*)d_in[16];
    float* out = (float*)d_out;

    // ---- workspace carve-up (≈225 MB total) ----
    char* ws = (char*)d_ws;
    size_t off = 0;
    auto alloc = [&](size_t bytes) -> void* {
        void* p = ws + off;
        off = (off + bytes + 255) & ~(size_t)255;
        return p;
    };
    float* resid  = (float*)alloc((size_t)cT * cD * 4);
    unsigned short* xnbf  = (unsigned short*)alloc((size_t)cT * cD * 2);
    float* skip   = (float*)alloc((size_t)cT * cE * 4);
    float* xin    = (float*)alloc((size_t)cT * cE * 4);
    float* xs     = (float*)alloc((size_t)cT * cE * 4);
    unsigned short* xsbf  = (unsigned short*)alloc((size_t)cT * cE * 2);
    float* d1     = (float*)alloc((size_t)cT * cDd * 4);
    unsigned short* d1bf  = (unsigned short*)alloc((size_t)cT * cDd * 2);
    float* delta  = (float*)alloc((size_t)cT * cE * 4);
    float* Bt     = (float*)alloc((size_t)cT * cN * 4);
    float* Ct     = (float*)alloc((size_t)cT * cN * 4);
    float* yv     = (float*)alloc((size_t)cT * cE * 4);
    unsigned short* zbf   = (unsigned short*)alloc((size_t)cT * cE * 2);
    unsigned short* wskip = (unsigned short*)alloc((size_t)cE * cD * 2);
    unsigned short* win   = (unsigned short*)alloc((size_t)cE * cD * 2);
    unsigned short* wout  = (unsigned short*)alloc((size_t)cD * cE * 2);
    unsigned short* wd1   = (unsigned short*)alloc((size_t)cDd * cE * 2);
    unsigned short* wd2   = (unsigned short*)alloc((size_t)cE * cDd * 2);
    unsigned short* wB    = (unsigned short*)alloc((size_t)cN * cE * 2);
    unsigned short* wC    = (unsigned short*)alloc((size_t)cN * cE * 2);
    unsigned short* wlm   = (unsigned short*)alloc((size_t)cV * cD * 2);  // 103 MB, fits L2

    const int TPB = 256;
    auto blks = [](long n) { return (int)((n + 255) / 256); };

    // 1) embedding gather
    embed_k<<<blks((long)cT * cD), TPB, 0, stream>>>(ids, emb, resid);

    for (int li = 0; li < cNL; ++li) {
        const float* nw  = l_norm_w + (long)li * cD;
        const float* swp = l_skip_w + (long)li * cE * cD;
        const float* iwp = l_in_w   + (long)li * cE * cD;
        const float* cwp = l_conv_w + (long)li * cE * cDc;
        const float* cbp = l_conv_b + (long)li * cE;
        const float* w1p = l_Wd1    + (long)li * cDd * cE;
        const float* w2p = l_Wd2_w  + (long)li * cE * cDd;
        const float* b2p = l_Wd2_b  + (long)li * cE;
        const float* wBp = l_WB     + (long)li * cN * cE;
        const float* wCp = l_WC     + (long)li * cN * cE;
        const float* alp = l_A_log  + (long)li * cE * cN;
        const float* wDp = l_WD     + (long)li * cE;
        const float* owp = l_out_w  + (long)li * cD * cE;

        // weight conversions (bf16)
        cvt_bf16_k<<<blks((long)cE * cD), TPB, 0, stream>>>(swp, wskip, cE * cD);
        cvt_bf16_k<<<blks((long)cE * cD), TPB, 0, stream>>>(iwp, win,  cE * cD);
        cvt_bf16_k<<<blks((long)cD * cE), TPB, 0, stream>>>(owp, wout, cD * cE);
        cvt_bf16_k<<<blks((long)cDd * cE), TPB, 0, stream>>>(w1p, wd1, cDd * cE);
        cvt_bf16_k<<<blks((long)cE * cDd), TPB, 0, stream>>>(w2p, wd2, cE * cDd);
        cvt_bf16_k<<<blks((long)cN * cE), TPB, 0, stream>>>(wBp, wB, cN * cE);
        cvt_bf16_k<<<blks((long)cN * cE), TPB, 0, stream>>>(wCp, wC, cN * cE);

        // rmsnorm -> xnbf
        rmsnorm_k<<<cT, TPB, 0, stream>>>(resid, nw, xnbf);

        // skip = xn @ skip_w^T ; x = xn @ in_w^T   [T,2048] K=1024
        {
            dim3 g((cT + 63) / 64, (cE + 255) / 256);
            wmma_gemm_bf16<<<g, TPB, 0, stream>>>(xnbf, wskip, skip, cT, cE, cD, cE, 0);
            wmma_gemm_bf16<<<g, TPB, 0, stream>>>(xnbf, win,  xin,  cT, cE, cD, cE, 0);
        }

        // depthwise causal conv + SiLU
        conv_silu_k<<<blks((long)cT * cE), TPB, 0, stream>>>(xin, cwp, cbp, xs, xsbf);

        // delta = softplus((xs @ Wd1^T) @ Wd2^T + b)
        {
            dim3 g1((cT + 63) / 64, (cDd + 255) / 256);
            wmma_gemm_bf16<<<g1, TPB, 0, stream>>>(xsbf, wd1, d1, cT, cDd, cE, cDd, 0);
            cvt_bf16_k<<<blks((long)cT * cDd), TPB, 0, stream>>>(d1, d1bf, cT * cDd);
            dim3 g2((cT + 63) / 64, (cE + 255) / 256);
            wmma_gemm_bf16<<<g2, TPB, 0, stream>>>(d1bf, wd2, delta, cT, cE, cDd, cE, 0);
            bias_softplus_k<<<blks((long)cT * cE), TPB, 0, stream>>>(delta, b2p);
        }

        // Bt = xs @ WB^T ; Ct = xs @ WC^T   [T,16] K=2048
        {
            dim3 g((cT + 63) / 64, 1);
            wmma_gemm_bf16<<<g, TPB, 0, stream>>>(xsbf, wB, Bt, cT, cN, cE, cN, 0);
            wmma_gemm_bf16<<<g, TPB, 0, stream>>>(xsbf, wC, Ct, cT, cN, cE, cN, 0);
        }

        // selective scan (recurrence over L)
        scan_k<<<blks((long)cB * cE), TPB, 0, stream>>>(delta, xs, Bt, Ct, alp, yv);

        // gate + skip
        zgate_k<<<blks((long)cT * cE), TPB, 0, stream>>>(yv, xs, skip, wDp, zbf);

        // resid += z @ out_w^T  [T,1024] K=2048, accumulate
        {
            dim3 g((cT + 63) / 64, (cD + 255) / 256);
            wmma_gemm_bf16<<<g, TPB, 0, stream>>>(zbf, wout, resid, cT, cD, cE, cD, 1);
        }
    }

    // final rmsnorm -> xnbf, convert LM head, big GEMM into d_out
    rmsnorm_k<<<cT, TPB, 0, stream>>>(resid, norm_w, xnbf);
    cvt_bf16_k<<<blks((long)cV * cD), TPB, 0, stream>>>(lm_head, wlm, cV * cD);
    {
        dim3 g((cT + 63) / 64, (cV + 255) / 256);
        wmma_gemm_bf16<<<g, TPB, 0, stream>>>(xnbf, wlm, out, cT, cV, cD, cV, 0);
    }
    (void)in_sizes; (void)n_in; (void)out_size; (void)ws_size;
}
